// MPNEncoder_9337258902201
// MI455X (gfx1250) — matrix operand order repaired and checked by reading, not compiled, not run on hardware
//
#include <hip/hip_runtime.h>

#define N_ATOMS   60000
#define N_BONDS   120000
#define MAX_NB    6
#define HIDDEN    512
#define ATOM_FDIM 133
#define BOND_FDIM 147
#define DEPTH     5
#define N_MOLS    512

#define KB_PAD 160    // BOND_FDIM padded to multiple of 32
#define KO_PAD 672    // concat dim: 160 (atom feats padded) + 512 (hidden)
#define AIN_OFF 160   // a_message offset inside a_input
#define M_ATOM_PAD 60032  // N_ATOMS padded to multiple of 64

#define LDS_PITCH 40  // halves per staged A row (32 + 8 pad): 80B, 16B-aligned, conflict-free

typedef __attribute__((ext_vector_type(16))) __bf16 v16bf;
typedef __attribute__((ext_vector_type(8)))  float  v8f;

union Frag16 {
  v16bf v;
  uint4 q[2];
  unsigned short s[16];
};

__device__ __forceinline__ float bf2f(unsigned short h) {
  union { unsigned int u; float f; } c; c.u = ((unsigned int)h) << 16; return c.f;
}
__device__ __forceinline__ unsigned short f2bf(float f) {
  union { float f; unsigned int u; } c; c.f = f;
  unsigned int u = c.u + 0x7FFFu + ((c.u >> 16) & 1u);
  return (unsigned short)(u >> 16);
}

// ---------------- CDNA5 async-to-LDS helpers (compile-safe guards) -------------

#if __has_builtin(__builtin_amdgcn_global_load_async_to_lds_b128)
#define HAS_ASYNC_LDS 1
#else
#define HAS_ASYNC_LDS 0
#endif

#if HAS_ASYNC_LDS
typedef __attribute__((__vector_size__(4 * sizeof(int)))) int v4i;
typedef v4i __attribute__((address_space(1)))* v4i_gptr;
typedef v4i __attribute__((address_space(3)))* v4i_lptr;
#endif

__device__ __forceinline__ void async_fence() {
#if HAS_ASYNC_LDS
#if __has_builtin(__builtin_amdgcn_s_wait_asynccnt)
  __builtin_amdgcn_s_wait_asynccnt(0);
#else
  asm volatile("s_wait_asynccnt 0x0" ::: "memory");
#endif
#endif
}

__device__ __forceinline__ void copy16_to_lds(const unsigned short* gsrc, unsigned short* ldst) {
#if HAS_ASYNC_LDS
  __builtin_amdgcn_global_load_async_to_lds_b128(
      (v4i_gptr)(unsigned long long)gsrc,
      (v4i_lptr)(unsigned int)(unsigned long long)ldst,
      0, 0);
#else
  *(uint4*)ldst = *(const uint4*)gsrc;
#endif
}

// ---------------- conversion / padding kernels ----------------

__global__ void k_cvt_pad(const float* __restrict__ src, unsigned short* __restrict__ dst,
                          int rows, int cs, int cd) {
  long t = (long)blockIdx.x * blockDim.x + threadIdx.x;
  long total = (long)rows * cd;
  if (t >= total) return;
  int r = (int)(t / cd), c = (int)(t % cd);
  dst[t] = (c < cs) ? f2bf(src[(long)r * cs + c]) : (unsigned short)0;
}

// W_o [512,645] -> [512,672]: cols 0..132 keep, 133..159 zero, 160..671 <- 133..644
__global__ void k_cvt_wo(const float* __restrict__ src, unsigned short* __restrict__ dst) {
  long t = (long)blockIdx.x * blockDim.x + threadIdx.x;
  long total = (long)HIDDEN * KO_PAD;
  if (t >= total) return;
  int r = (int)(t / KO_PAD), c = (int)(t % KO_PAD);
  unsigned short v = 0;
  if (c < ATOM_FDIM)       v = f2bf(src[(long)r * (ATOM_FDIM + HIDDEN) + c]);
  else if (c >= AIN_OFF)   v = f2bf(src[(long)r * (ATOM_FDIM + HIDDEN) + (c - (AIN_OFF - ATOM_FDIM))]);
  dst[t] = v;
}

// a_input [60032,672]: f_atoms | zeros | a_message, zero rows beyond N_ATOMS
__global__ void k_build_ain(const float* __restrict__ f_atoms,
                            const unsigned short* __restrict__ amsg,
                            unsigned short* __restrict__ ain) {
  long t = (long)blockIdx.x * blockDim.x + threadIdx.x;
  long total = (long)M_ATOM_PAD * KO_PAD;
  if (t >= total) return;
  int r = (int)(t / KO_PAD), c = (int)(t % KO_PAD);
  unsigned short v = 0;
  if (r < N_ATOMS) {
    if (c < ATOM_FDIM)      v = f2bf(f_atoms[(long)r * ATOM_FDIM + c]);
    else if (c >= AIN_OFF)  v = amsg[(long)r * HIDDEN + (c - AIN_OFF)];
  }
  ain[t] = v;
}

// ---------------- neighbor-sum: a_msg[a] = sum_j msg[a2b[a][j]] ----------------

__global__ void k_sum(const unsigned short* __restrict__ msg,
                      const long long* __restrict__ a2b,
                      unsigned short* __restrict__ amsg) {
  long t = (long)blockIdx.x * blockDim.x + threadIdx.x;
  if (t >= (long)N_ATOMS * (HIDDEN / 8)) return;
  int a = (int)(t >> 6);
  int c = ((int)t & 63) << 3;
  const long long* nb = a2b + (long)a * MAX_NB;
  float s[8];
#pragma unroll
  for (int i = 0; i < 8; ++i) s[i] = 0.f;
#pragma unroll
  for (int j = 0; j < MAX_NB; ++j) {
    const uint4 q = *(const uint4*)(msg + (size_t)nb[j] * HIDDEN + c);
    const unsigned short* u = (const unsigned short*)&q;
#pragma unroll
    for (int i = 0; i < 8; ++i) s[i] += bf2f(u[i]);
  }
  uint4 o; unsigned short* ou = (unsigned short*)&o;
#pragma unroll
  for (int i = 0; i < 8; ++i) ou[i] = f2bf(s[i]);
  *(uint4*)(amsg + (size_t)a * HIDDEN + c) = o;
}

// ---------------- WMMA GEMM ----------------------------------------------------
// Block: 256 threads (8 waves). Block tile 64(M) x 512(N); wave tile 64x64.
// A tile (64x32 bf16) staged in LDS, double-buffered; per K-step each wave does
// 16 v_wmma_f32_16x16x32_bf16 fed by 8 global b128 (B) + 8 ds b128 (A).
// MODE 0: C = A@W^T          -> store raw bf16 (oinp) and relu bf16 (omsg)
// MODE 1: A = amsg[b2a]-msg[b2revb] fused into LDS staging; C += inp; relu -> omsg
// MODE 2: C += b_o; relu; atomicAdd per-molecule sums (rows >= Mvalid skipped)

template <int MODE>
__device__ __forceinline__ void stage_tile(unsigned short* dst,
                                           const unsigned short* __restrict__ Abase, int lda,
                                           int k0, int m0,
                                           const unsigned short* __restrict__ amsg,
                                           const unsigned short* __restrict__ msgc,
                                           const long long* __restrict__ b2a,
                                           const long long* __restrict__ b2revb) {
  const int t  = threadIdx.x;
  const int r  = t >> 2;             // 0..63
  const int ch = (t & 3) << 3;       // 0,8,16,24 halves
  unsigned short* d = dst + r * LDS_PITCH + ch;
  if (MODE == 1) {
    const int bond = m0 + r;
    const unsigned short* p0 = amsg + (size_t)b2a[bond] * HIDDEN + k0 + ch;
    const unsigned short* p1 = msgc + (size_t)b2revb[bond] * HIDDEN + k0 + ch;
    uint4 x = *(const uint4*)p0;
    uint4 y = *(const uint4*)p1;
    const unsigned short* xs = (const unsigned short*)&x;
    const unsigned short* ys = (const unsigned short*)&y;
    uint4 o; unsigned short* os = (unsigned short*)&o;
#pragma unroll
    for (int i = 0; i < 8; ++i) os[i] = f2bf(bf2f(xs[i]) - bf2f(ys[i]));
    *(uint4*)d = o;
  } else {
    copy16_to_lds(Abase + (size_t)r * lda + k0 + ch, d);
  }
}

template <int MODE>
__global__ __launch_bounds__(256, 1)
void k_gemm(const unsigned short* __restrict__ A, int lda,
            const unsigned short* __restrict__ W, int ldw, int K, int Mvalid,
            const unsigned short* __restrict__ inp,
            unsigned short* __restrict__ omsg,
            unsigned short* __restrict__ oinp,
            const unsigned short* __restrict__ amsg,
            const unsigned short* __restrict__ msgc,
            const long long* __restrict__ b2a,
            const long long* __restrict__ b2revb,
            const float* __restrict__ bo,
            const long long* __restrict__ mol_ids,
            float* __restrict__ sums) {
  __shared__ alignas(16) unsigned short smem[2][64 * LDS_PITCH];

  const int lane = threadIdx.x & 31;
  const int wv   = threadIdx.x >> 5;
  const int m0   = blockIdx.x << 6;   // 64 rows per block
  const int n0   = wv << 6;           // 64 cols per wave
  const int lr   = lane & 15;
  const int hk   = lane >> 4;

  const unsigned short* Abase = (MODE == 1) ? nullptr : (A + (size_t)m0 * lda);

  const v8f vz = {0.f, 0.f, 0.f, 0.f, 0.f, 0.f, 0.f, 0.f};
  v8f acc[4][4];
#pragma unroll
  for (int mi = 0; mi < 4; ++mi)
#pragma unroll
    for (int ni = 0; ni < 4; ++ni) acc[mi][ni] = vz;

  stage_tile<MODE>(smem[0], Abase, lda, 0, m0, amsg, msgc, b2a, b2revb);
  int cur = 0;

  for (int k0 = 0; k0 < K; k0 += 32) {
    async_fence();
    __syncthreads();
    if (k0 + 32 < K)
      stage_tile<MODE>(smem[cur ^ 1], Abase, lda, k0 + 32, m0, amsg, msgc, b2a, b2revb);

    // B fragments: lane holds weight row n0+ni*16+lr, K-halves 8hk..8hk+7 and 16+8hk..16+8hk+7
    const size_t s16 = (size_t)ldw << 4;
    const unsigned short* wp = W + (size_t)(n0 + lr) * ldw + k0 + (hk << 3);
    Frag16 b0, b1, b2, b3;
    b0.q[0] = *(const uint4*)(wp);           b0.q[1] = *(const uint4*)(wp + 16);
    b1.q[0] = *(const uint4*)(wp + s16);     b1.q[1] = *(const uint4*)(wp + s16 + 16);
    b2.q[0] = *(const uint4*)(wp + 2 * s16); b2.q[1] = *(const uint4*)(wp + 2 * s16 + 16);
    b3.q[0] = *(const uint4*)(wp + 3 * s16); b3.q[1] = *(const uint4*)(wp + 3 * s16 + 16);

    const unsigned short* bufc = smem[cur];
#pragma unroll
    for (int mi = 0; mi < 4; ++mi) {
      const unsigned short* ap = bufc + (mi * 16 + lr) * LDS_PITCH + (hk << 3);
      Frag16 a;
      a.q[0] = *(const uint4*)(ap);
      a.q[1] = *(const uint4*)(ap + 16);
      acc[mi][0] = __builtin_amdgcn_wmma_f32_16x16x32_bf16(false, a.v, false, b0.v, (short)0, acc[mi][0], false, false);
      acc[mi][1] = __builtin_amdgcn_wmma_f32_16x16x32_bf16(false, a.v, false, b1.v, (short)0, acc[mi][1], false, false);
      acc[mi][2] = __builtin_amdgcn_wmma_f32_16x16x32_bf16(false, a.v, false, b2.v, (short)0, acc[mi][2], false, false);
      acc[mi][3] = __builtin_amdgcn_wmma_f32_16x16x32_bf16(false, a.v, false, b3.v, (short)0, acc[mi][3], false, false);
    }
    cur ^= 1;
  }

#pragma unroll
  for (int mi = 0; mi < 4; ++mi) {
#pragma unroll
    for (int ni = 0; ni < 4; ++ni) {
      const int col = n0 + (ni << 4) + lr;
#pragma unroll
      for (int r = 0; r < 8; ++r) {
        const int mr = m0 + (mi << 4) + (hk << 3) + r;
        const size_t idx = (size_t)mr * HIDDEN + col;
        float v = acc[mi][ni][r];
        if (MODE == 0) {
          oinp[idx] = f2bf(v);
          omsg[idx] = f2bf(v > 0.f ? v : 0.f);
        } else if (MODE == 1) {
          v += bf2f(inp[idx]);
          omsg[idx] = f2bf(v > 0.f ? v : 0.f);
        } else {
          if (mr < Mvalid) {
            v += bo[col];
            v = v > 0.f ? v : 0.f;
            const long long mol = mol_ids[mr];
            atomicAdd(&sums[(size_t)mol * HIDDEN + col], v);
          }
        }
      }
    }
  }
}

// ---------------- pooling helpers ----------------

__global__ void k_zero(float* __restrict__ p, long n) {
  long t = (long)blockIdx.x * blockDim.x + threadIdx.x;
  if (t < n) p[t] = 0.f;
}

__global__ void k_count(const long long* __restrict__ mol_ids, float* __restrict__ cnts) {
  long t = (long)blockIdx.x * blockDim.x + threadIdx.x;
  if (t < N_ATOMS) atomicAdd(&cnts[mol_ids[t]], 1.0f);
}

__global__ void k_final(const float* __restrict__ sums, const float* __restrict__ cnts,
                        float* __restrict__ out) {
  long t = (long)blockIdx.x * blockDim.x + threadIdx.x;
  if (t >= (long)N_MOLS * HIDDEN) return;
  out[t] = sums[t] / fmaxf(cnts[t >> 9], 1.0f);
}

// ---------------- launch ----------------

extern "C" void kernel_launch(void* const* d_in, const int* in_sizes, int n_in,
                              void* d_out, int out_size, void* d_ws, size_t ws_size,
                              hipStream_t stream) {
  (void)in_sizes; (void)n_in; (void)out_size; (void)ws_size;
  const float*     f_atoms = (const float*)d_in[0];
  const float*     f_bonds = (const float*)d_in[1];
  const float*     W_i     = (const float*)d_in[2];
  const float*     W_h     = (const float*)d_in[3];
  const float*     W_o     = (const float*)d_in[4];
  const float*     b_o     = (const float*)d_in[5];
  const long long* a2b     = (const long long*)d_in[6];
  const long long* b2a     = (const long long*)d_in[7];
  const long long* b2revb  = (const long long*)d_in[8];
  const long long* mol_ids = (const long long*)d_in[9];
  float* out = (float*)d_out;

  char* w = (char*)d_ws;
  auto alloc = [&](size_t bytes) -> char* {
    char* p = w; w += (bytes + 255) & ~(size_t)255; return p;
  };
  unsigned short* fb   = (unsigned short*)alloc((size_t)N_BONDS * KB_PAD * 2);
  unsigned short* Wi   = (unsigned short*)alloc((size_t)HIDDEN * KB_PAD * 2);
  unsigned short* Wh   = (unsigned short*)alloc((size_t)HIDDEN * HIDDEN * 2);
  unsigned short* Wo   = (unsigned short*)alloc((size_t)HIDDEN * KO_PAD * 2);
  unsigned short* inp  = (unsigned short*)alloc((size_t)N_BONDS * HIDDEN * 2);
  unsigned short* msgA = (unsigned short*)alloc((size_t)N_BONDS * HIDDEN * 2);
  unsigned short* msgB = (unsigned short*)alloc((size_t)N_BONDS * HIDDEN * 2);
  unsigned short* amsg = (unsigned short*)alloc((size_t)N_ATOMS * HIDDEN * 2);
  unsigned short* ain  = (unsigned short*)alloc((size_t)M_ATOM_PAD * KO_PAD * 2);
  float* sums = (float*)alloc(((size_t)N_MOLS * HIDDEN + N_MOLS) * 4);
  float* cnts = sums + (size_t)N_MOLS * HIDDEN;

  const dim3 blk(256);
  auto g1 = [](long n) { return dim3((unsigned)((n + 255) / 256)); };

  k_cvt_pad<<<g1((long)N_BONDS * KB_PAD), blk, 0, stream>>>(f_bonds, fb, N_BONDS, BOND_FDIM, KB_PAD);
  k_cvt_pad<<<g1((long)HIDDEN * KB_PAD), blk, 0, stream>>>(W_i, Wi, HIDDEN, BOND_FDIM, KB_PAD);
  k_cvt_pad<<<g1((long)HIDDEN * HIDDEN), blk, 0, stream>>>(W_h, Wh, HIDDEN, HIDDEN, HIDDEN);
  k_cvt_wo<<<g1((long)HIDDEN * KO_PAD), blk, 0, stream>>>(W_o, Wo);
  k_zero<<<g1((long)N_MOLS * HIDDEN + N_MOLS), blk, 0, stream>>>(sums, (long)N_MOLS * HIDDEN + N_MOLS);

  // inp = f_bonds @ W_i^T ; msg = relu(inp)
  k_gemm<0><<<dim3(N_BONDS / 64), blk, 0, stream>>>(
      fb, KB_PAD, Wi, KB_PAD, KB_PAD, N_BONDS,
      nullptr, msgA, inp, nullptr, nullptr, nullptr, nullptr, nullptr, nullptr, nullptr);

  unsigned short* mc = msgA;
  unsigned short* mn = msgB;
  for (int d = 0; d < DEPTH - 1; ++d) {
    k_sum<<<g1((long)N_ATOMS * (HIDDEN / 8)), blk, 0, stream>>>(mc, a2b, amsg);
    k_gemm<1><<<dim3(N_BONDS / 64), blk, 0, stream>>>(
        nullptr, 0, Wh, HIDDEN, HIDDEN, N_BONDS,
        inp, mn, nullptr, amsg, mc, b2a, b2revb, nullptr, nullptr, nullptr);
    unsigned short* t = mc; mc = mn; mn = t;
  }
  k_sum<<<g1((long)N_ATOMS * (HIDDEN / 8)), blk, 0, stream>>>(mc, a2b, amsg);
  k_build_ain<<<g1((long)M_ATOM_PAD * KO_PAD), blk, 0, stream>>>(f_atoms, amsg, ain);
  k_count<<<g1((long)N_ATOMS), blk, 0, stream>>>(mol_ids, cnts);

  // atom_hiddens = relu(a_input @ W_o^T + b_o), fused molecule-sum via atomics
  k_gemm<2><<<dim3(M_ATOM_PAD / 64), blk, 0, stream>>>(
      ain, KO_PAD, Wo, KO_PAD, KO_PAD, N_ATOMS,
      nullptr, nullptr, nullptr, nullptr, nullptr, nullptr, nullptr, b_o, mol_ids, sums);

  k_final<<<g1((long)N_MOLS * HIDDEN), blk, 0, stream>>>(sums, cnts, out);
}
